// Encoder_19559281066218
// MI455X (gfx1250) — matrix operand order repaired and checked by reading, not compiled, not run on hardware
//
#include <hip/hip_runtime.h>
#include <hip/hip_bf16.h>

typedef __attribute__((ext_vector_type(16))) __bf16 v16bf;
typedef __attribute__((ext_vector_type(8)))  __bf16 v8bf;
typedef __attribute__((ext_vector_type(8)))  float  v8f;
typedef int v4i_ __attribute__((vector_size(16)));

#define B_   8
#define T_   1024
#define C_   768
#define H_   12
#define HS_  64
#define DFF_ 3072
#define M_   (B_*T_)

#if defined(__AMDGCN__) && __has_builtin(__builtin_amdgcn_global_load_async_to_lds_b128) && __has_builtin(__builtin_amdgcn_s_wait_asynccnt)
#define HAVE_ASYNC_LDS 1
#else
#define HAVE_ASYNC_LDS 0
#endif

#if HAVE_ASYNC_LDS
__device__ __forceinline__ void async_cp16(const __bf16* g, __bf16* l) {
    __builtin_amdgcn_global_load_async_to_lds_b128(
        (__attribute__((address_space(1))) v4i_*)g,
        (__attribute__((address_space(3))) v4i_*)l,
        0, 0);
}
#endif

__device__ __forceinline__ __bf16 f2bf(float f) {
    unsigned u = __float_as_uint(f);
    u = u + 0x7FFFu + ((u >> 16) & 1u);          // round-to-nearest-even
    unsigned short s = (unsigned short)(u >> 16);
    return __builtin_bit_cast(__bf16, s);
}

__device__ __forceinline__ v8f zero8() {
    v8f z;
#pragma unroll
    for (int i = 0; i < 8; ++i) z[i] = 0.0f;
    return z;
}

// A-fragment: 16xK tile (row-major, K-step 32). ISA 16-bit A layout.
__device__ __forceinline__ v16bf load_a16(const __bf16* base, int ld) {
    int lane = threadIdx.x & 31;
    int row  = lane & 15;
    int kh   = (lane >> 4) << 3;                 // 0 or 8
    const __bf16* p = base + row * ld + kh;
    v8bf lo = *(const v8bf*)p;
    v8bf hi = *(const v8bf*)(p + 16);
    return __builtin_shufflevector(lo, hi, 0,1,2,3,4,5,6,7,8,9,10,11,12,13,14,15);
}

// B-fragment from Bt (B transposed, N x K row-major). ISA 16-bit B layout.
__device__ __forceinline__ v16bf load_b16(const __bf16* base, int ld) {
    int lane = threadIdx.x & 31;
    int n    = lane & 15;
    int kb   = (lane >> 4) << 3;                 // 0 or 8
    const __bf16* p = base + n * ld + kb;
    v8bf lo = *(const v8bf*)p;
    v8bf hi = *(const v8bf*)(p + 16);
    return __builtin_shufflevector(lo, hi, 0,1,2,3,4,5,6,7,8,9,10,11,12,13,14,15);
}

__device__ __forceinline__ v8f wmma_bf(v16bf a, v16bf b, v8f c) {
    return __builtin_amdgcn_wmma_f32_16x16x32_bf16(false, a, false, b, (short)0, c, false, false);
}

// ---------------- weight prep ----------------
__global__ void k_transpose_bf16(const float* __restrict__ src, __bf16* __restrict__ dst,
                                 int batch, int R, int Cc) {
    long long total = (long long)batch * R * Cc;
    long long idx = (long long)blockIdx.x * blockDim.x + threadIdx.x;
    if (idx >= total) return;
    int c = (int)(idx % Cc);
    long long t = idx / Cc;
    int r = (int)(t % R);
    int b = (int)(t / R);
    dst[((long long)b * Cc + c) * R + r] = f2bf(src[idx]);
}

__global__ void k_concat3(const float* a, const float* b, const float* c, float* dst, int n) {
    int i = blockIdx.x * blockDim.x + threadIdx.x;
    if (i < n) dst[i] = a[i];
    else if (i < 2 * n) dst[i] = b[i - n];
    else if (i < 3 * n) dst[i] = c[i - 2 * n];
}

// ---------------- layernorm ----------------
__global__ __launch_bounds__(256) void k_ln(const float* __restrict__ x,
                                            const float* __restrict__ g,
                                            const float* __restrict__ be,
                                            float* __restrict__ outf,
                                            __bf16* __restrict__ outb, int C) {
    int row = blockIdx.x;
    const float* xr = x + (size_t)row * C;
    float s = 0.f, ss = 0.f;
    for (int i = threadIdx.x; i < C; i += blockDim.x) { float v = xr[i]; s += v; ss += v * v; }
    __shared__ float red[64];
#pragma unroll
    for (int o = 16; o > 0; o >>= 1) { s += __shfl_xor(s, o, 32); ss += __shfl_xor(ss, o, 32); }
    int wid = threadIdx.x >> 5, ln = threadIdx.x & 31;
    if (ln == 0) { red[wid] = s; red[32 + wid] = ss; }
    __syncthreads();
    if (wid == 0) {
        float a  = (ln < 8) ? red[ln] : 0.f;
        float c2 = (ln < 8) ? red[32 + ln] : 0.f;
#pragma unroll
        for (int o = 4; o > 0; o >>= 1) { a += __shfl_xor(a, o, 8); c2 += __shfl_xor(c2, o, 8); }
        if (ln == 0) { red[0] = a; red[1] = c2; }
    }
    __syncthreads();
    float mu  = red[0] / (float)C;
    float var = red[1] / (float)C - mu * mu;
    float inv = rsqrtf(var + 1e-5f);
    for (int i = threadIdx.x; i < C; i += blockDim.x) {
        float v = (xr[i] - mu) * inv * g[i] + be[i];
        outf[(size_t)row * C + i] = v;
        outb[(size_t)row * C + i] = f2bf(v);
    }
}

// ---------------- bf16 WMMA GEMM with double-buffered LDS B-tile ----------------
// C[M,N] = A[M,K] * Bt[N,K]^T ; wave = 16x64 strip, block = 128x64
#define BSTR 40   // padded LDS row stride (halves) for 32-half B rows
__global__ __launch_bounds__(256) void k_gemm(const __bf16* __restrict__ A, int lda,
                                              const __bf16* __restrict__ Bt, int ldb,
                                              const float* __restrict__ bias,
                                              const float* __restrict__ resid,
                                              float* __restrict__ outF,
                                              __bf16* __restrict__ outB,
                                              int ldo, int K, int relu) {
    __shared__ __align__(16) __bf16 bts[2][64 * BSTR];
    int tid = threadIdx.x;
    int lane = tid & 31, wid = tid >> 5;
    int m0 = blockIdx.y * 128 + wid * 16;
    int n0 = blockIdx.x * 64;

    // cooperative B fill: 64 rows x 32 halves = 256 threads x 16B
    int brow = tid >> 2;
    int bch  = (tid & 3) << 3;
    const __bf16* bsrc = Bt + (long long)(n0 + brow) * ldb + bch;
    __bf16* bd0 = &bts[0][brow * BSTR + bch];
    __bf16* bd1 = &bts[1][brow * BSTR + bch];

#if HAVE_ASYNC_LDS
    async_cp16(bsrc, bd0);
    __builtin_amdgcn_s_wait_asynccnt(0);
#else
    *(v8bf*)bd0 = *(const v8bf*)bsrc;
#endif
    __syncthreads();

    v8f acc0 = zero8(), acc1 = zero8(), acc2 = zero8(), acc3 = zero8();
    int nk = K >> 5;
    for (int kt = 0; kt < nk; ++kt) {
        int kk = kt << 5;
        const __bf16* bcur = &bts[kt & 1][0];
        __bf16* bnxt = (kt & 1) ? bd0 : bd1;
#if HAVE_ASYNC_LDS
        if (kt + 1 < nk) async_cp16(bsrc + kk + 32, bnxt);
#else
        v8bf nreg;
        if (kt + 1 < nk) nreg = *(const v8bf*)(bsrc + kk + 32);
#endif
        v16bf a = load_a16(A + (long long)m0 * lda + kk, lda);
        acc0 = wmma_bf(a, load_b16(bcur +  0 * BSTR, BSTR), acc0);
        acc1 = wmma_bf(a, load_b16(bcur + 16 * BSTR, BSTR), acc1);
        acc2 = wmma_bf(a, load_b16(bcur + 32 * BSTR, BSTR), acc2);
        acc3 = wmma_bf(a, load_b16(bcur + 48 * BSTR, BSTR), acc3);
#if HAVE_ASYNC_LDS
        if (kt + 1 < nk) __builtin_amdgcn_s_wait_asynccnt(0);
#else
        if (kt + 1 < nk) *(v8bf*)bnxt = nreg;
#endif
        __syncthreads();
    }

    int half8 = (lane >> 4) << 3;
    int cb = n0 + (lane & 15);
    v8f* accs[4] = { &acc0, &acc1, &acc2, &acc3 };
#pragma unroll
    for (int j = 0; j < 4; ++j) {
        int col = cb + j * 16;
        float bv = bias ? bias[col] : 0.f;
#pragma unroll
        for (int r = 0; r < 8; ++r) {
            long long row = m0 + r + half8;
            float v = (*accs[j])[r] + bv;
            if (relu) v = fmaxf(v, 0.f);
            if (resid) v += resid[row * ldo + col];
            if (outF) outF[row * ldo + col] = v;
            else      outB[row * ldo + col] = f2bf(v);
        }
    }
}

// ---------------- split fused qkv into q, k, v^T ----------------
__global__ void k_split_qkv(const __bf16* __restrict__ qkv,
                            __bf16* __restrict__ q, __bf16* __restrict__ kk,
                            __bf16* __restrict__ vt) {
    long long total = (long long)M_ * 3 * C_;
    long long idx = (long long)blockIdx.x * blockDim.x + threadIdx.x;
    if (idx >= total) return;
    int col = (int)(idx % (3 * C_));
    long long row = idx / (3 * C_);
    int b = (int)(row / T_), t = (int)(row % T_);
    int sect = col / C_, w = col % C_;
    int h = w / HS_, d = w % HS_;
    __bf16 val = qkv[idx];
    long long bh = (long long)b * H_ + h;
    if (sect == 0)      q [(bh * T_ + t) * HS_ + d] = val;
    else if (sect == 1) kk[(bh * T_ + t) * HS_ + d] = val;
    else                vt[(bh * HS_ + d) * T_ + t] = val;
}

// ---------------- flash attention, K/V staged in double-buffered LDS ----------------
// grid = (T/128, B*H); block = 8 waves sharing one (b,h), wave = one 16-row q tile
#define KSTR 72   // 64-half K rows, padded
#define VSTR 40   // 32-half V rows, padded
__global__ __launch_bounds__(256) void k_attn(const __bf16* __restrict__ q,
                                              const __bf16* __restrict__ kmat,
                                              const __bf16* __restrict__ vt,
                                              __bf16* __restrict__ obf) {
    __shared__ __align__(16) __bf16 kts[2][32 * KSTR];
    __shared__ __align__(16) __bf16 vts[2][64 * VSTR];
    __shared__ __align__(16) __bf16 pbuf[8][16 * 32];
    int tid = threadIdx.x;
    int lane = tid & 31, wid = tid >> 5;
    int bh = blockIdx.y;
    int b = bh / H_, h = bh % H_;
    int q0 = blockIdx.x * 128 + wid * 16;
    const float scale = 0.03608439182435161f;    // 1/sqrt(768) (ref scales by sqrt(C))

    // cooperative fill indices: K tile 32x64 halves, V tile 64x32 halves
    int krow = tid >> 3, kch = (tid & 7) << 3;
    int vrow = tid >> 2, vch = (tid & 3) << 3;
    const __bf16* ksrc = kmat + ((long long)bh * T_ + krow) * HS_ + kch;
    const __bf16* vsrc = vt + ((long long)bh * HS_ + vrow) * T_ + vch;
    __bf16* kd0 = &kts[0][krow * KSTR + kch];
    __bf16* kd1 = &kts[1][krow * KSTR + kch];
    __bf16* vd0 = &vts[0][vrow * VSTR + vch];
    __bf16* vd1 = &vts[1][vrow * VSTR + vch];

    const __bf16* qp = q + ((long long)bh * T_ + q0) * HS_;
    v16bf aq0 = load_a16(qp, HS_);
    v16bf aq1 = load_a16(qp + 32, HS_);

#if HAVE_ASYNC_LDS
    async_cp16(ksrc, kd0);
    async_cp16(vsrc, vd0);
    __builtin_amdgcn_s_wait_asynccnt(0);
#else
    *(v8bf*)kd0 = *(const v8bf*)ksrc;
    *(v8bf*)vd0 = *(const v8bf*)vsrc;
#endif
    __syncthreads();

    v8f o0 = zero8(), o1 = zero8(), o2 = zero8(), o3 = zero8();
    float m[8], l[8];
#pragma unroll
    for (int r = 0; r < 8; ++r) { m[r] = -1e30f; l[r] = 0.f; }

    const int nst = T_ / 32;
    for (int st = 0; st < nst; ++st) {
        int s0 = st << 5;
        int cur = st & 1;
        const __bf16* kc = &kts[cur][0];
        const __bf16* vc = &vts[cur][0];
        __bf16* knx = cur ? kd0 : kd1;
        __bf16* vnx = cur ? vd0 : vd1;
#if HAVE_ASYNC_LDS
        if (st + 1 < nst) { async_cp16(ksrc + (s0 + 32) * HS_, knx); async_cp16(vsrc + s0 + 32, vnx); }
#else
        v8bf kreg, vreg;
        if (st + 1 < nst) { kreg = *(const v8bf*)(ksrc + (s0 + 32) * HS_); vreg = *(const v8bf*)(vsrc + s0 + 32); }
#endif
        v8f sc0 = zero8(), sc1 = zero8();
        sc0 = wmma_bf(aq0, load_b16(kc, KSTR), sc0);
        sc0 = wmma_bf(aq1, load_b16(kc + 32, KSTR), sc0);
        sc1 = wmma_bf(aq0, load_b16(kc + 16 * KSTR, KSTR), sc1);
        sc1 = wmma_bf(aq1, load_b16(kc + 16 * KSTR + 32, KSTR), sc1);

        float al[8];
#pragma unroll
        for (int r = 0; r < 8; ++r) {
            float x0 = sc0[r] * scale, x1 = sc1[r] * scale;
            sc0[r] = x0; sc1[r] = x1;
            float tm = fmaxf(x0, x1);
            tm = fmaxf(tm, __shfl_xor(tm, 1, 16));
            tm = fmaxf(tm, __shfl_xor(tm, 2, 16));
            tm = fmaxf(tm, __shfl_xor(tm, 4, 16));
            tm = fmaxf(tm, __shfl_xor(tm, 8, 16));
            float mn = fmaxf(m[r], tm);
            al[r] = __expf(m[r] - mn);
            m[r] = mn;
        }
#pragma unroll
        for (int r = 0; r < 8; ++r) {
            float p0 = __expf(sc0[r] - m[r]);
            float p1 = __expf(sc1[r] - m[r]);
            o0[r] *= al[r]; o1[r] *= al[r]; o2[r] *= al[r]; o3[r] *= al[r];
            float rs = p0 + p1;
            rs += __shfl_xor(rs, 1, 16);
            rs += __shfl_xor(rs, 2, 16);
            rs += __shfl_xor(rs, 4, 16);
            rs += __shfl_xor(rs, 8, 16);
            l[r] = l[r] * al[r] + rs;
            int prow = r + ((lane >> 4) << 3);
            pbuf[wid][prow * 32 + (lane & 15)]      = f2bf(p0);
            pbuf[wid][prow * 32 + 16 + (lane & 15)] = f2bf(p1);
        }
        asm volatile("s_wait_dscnt 0" ::: "memory");   // C-frag -> A-frag relayout via LDS
        v16bf pa = load_a16(&pbuf[wid][0], 32);
        o0 = wmma_bf(pa, load_b16(vc +  0 * VSTR, VSTR), o0);
        o1 = wmma_bf(pa, load_b16(vc + 16 * VSTR, VSTR), o1);
        o2 = wmma_bf(pa, load_b16(vc + 32 * VSTR, VSTR), o2);
        o3 = wmma_bf(pa, load_b16(vc + 48 * VSTR, VSTR), o3);
#if HAVE_ASYNC_LDS
        if (st + 1 < nst) __builtin_amdgcn_s_wait_asynccnt(0);
#else
        if (st + 1 < nst) { *(v8bf*)knx = kreg; *(v8bf*)vnx = vreg; }
#endif
        __syncthreads();
    }

    int half8 = (lane >> 4) << 3;
#pragma unroll
    for (int r = 0; r < 8; ++r) {
        float inv = 1.f / l[r];
        long long row = (long long)b * T_ + q0 + r + half8;
        int colb = h * HS_ + (lane & 15);
        obf[row * C_ + colb +  0] = f2bf(o0[r] * inv);
        obf[row * C_ + colb + 16] = f2bf(o1[r] * inv);
        obf[row * C_ + colb + 32] = f2bf(o2[r] * inv);
        obf[row * C_ + colb + 48] = f2bf(o3[r] * inv);
    }
}

extern "C" void kernel_launch(void* const* d_in, const int* in_sizes, int n_in,
                              void* d_out, int out_size, void* d_ws, size_t ws_size,
                              hipStream_t stream) {
    (void)in_sizes; (void)n_in; (void)out_size; (void)ws_size;
    const float* x   = (const float*)d_in[0];
    const float* Wq  = (const float*)d_in[1];
    const float* bq  = (const float*)d_in[2];
    const float* Wk  = (const float*)d_in[3];
    const float* bk  = (const float*)d_in[4];
    const float* Wv  = (const float*)d_in[5];
    const float* bv  = (const float*)d_in[6];
    const float* Wp  = (const float*)d_in[7];
    const float* bp  = (const float*)d_in[8];
    const float* W1  = (const float*)d_in[9];
    const float* b1  = (const float*)d_in[10];
    const float* W2  = (const float*)d_in[11];
    const float* b2  = (const float*)d_in[12];
    const float* g1  = (const float*)d_in[13];
    const float* be1 = (const float*)d_in[14];
    const float* g2  = (const float*)d_in[15];
    const float* be2 = (const float*)d_in[16];

    size_t off = 0;
    auto alloc = [&](size_t bytes) -> void* {
        void* p = (char*)d_ws + off;
        off += (bytes + 255) & ~(size_t)255;
        return p;
    };
    float*  h1f   = (float*) alloc((size_t)M_ * C_ * 4);
    __bf16* h1b   = (__bf16*)alloc((size_t)M_ * C_ * 2);
    __bf16* wqkv  = (__bf16*)alloc((size_t)3 * C_ * C_ * 2);
    __bf16* wpt   = (__bf16*)alloc((size_t)C_ * C_ * 2);
    __bf16* w1t   = (__bf16*)alloc((size_t)C_ * DFF_ * 2);
    __bf16* w2t   = (__bf16*)alloc((size_t)C_ * DFF_ * 2);
    float*  bqkv  = (float*) alloc((size_t)3 * C_ * 4);
    __bf16* qkv   = (__bf16*)alloc((size_t)M_ * 3 * C_ * 2);
    __bf16* qb    = (__bf16*)alloc((size_t)M_ * C_ * 2);
    __bf16* kb    = (__bf16*)alloc((size_t)M_ * C_ * 2);
    __bf16* vtb   = (__bf16*)alloc((size_t)M_ * C_ * 2);
    __bf16* ob    = (__bf16*)alloc((size_t)M_ * C_ * 2);
    float*  hattn = (float*) alloc((size_t)M_ * C_ * 4);
    float*  h2f   = (float*) alloc((size_t)M_ * C_ * 4);
    __bf16* h2b   = (__bf16*)alloc((size_t)M_ * C_ * 2);
    __bf16* act   = (__bf16*)alloc((size_t)M_ * DFF_ * 2);

    const int thr = 256;
    long long n = (long long)H_ * C_ * HS_;
    k_transpose_bf16<<<(unsigned)((n + thr - 1) / thr), thr, 0, stream>>>(Wq, wqkv,               H_, C_, HS_);
    k_transpose_bf16<<<(unsigned)((n + thr - 1) / thr), thr, 0, stream>>>(Wk, wqkv + C_ * C_,     H_, C_, HS_);
    k_transpose_bf16<<<(unsigned)((n + thr - 1) / thr), thr, 0, stream>>>(Wv, wqkv + 2 * C_ * C_, H_, C_, HS_);
    n = (long long)C_ * C_;
    k_transpose_bf16<<<(unsigned)((n + thr - 1) / thr), thr, 0, stream>>>(Wp, wpt, 1, C_, C_);
    n = (long long)C_ * DFF_;
    k_transpose_bf16<<<(unsigned)((n + thr - 1) / thr), thr, 0, stream>>>(W1, w1t, 1, C_, DFF_);
    k_transpose_bf16<<<(unsigned)((n + thr - 1) / thr), thr, 0, stream>>>(W2, w2t, 1, DFF_, C_);
    k_concat3<<<(3 * C_ + thr - 1) / thr, thr, 0, stream>>>(bq, bk, bv, bqkv, C_);

    k_ln<<<M_, thr, 0, stream>>>(x, g1, be1, h1f, h1b, C_);

    dim3 gq(3 * C_ / 64, M_ / 128);
    k_gemm<<<gq, thr, 0, stream>>>(h1b, C_, wqkv, C_, bqkv, nullptr, nullptr, qkv, 3 * C_, C_, 0);

    long long tot = (long long)M_ * 3 * C_;
    k_split_qkv<<<(unsigned)((tot + thr - 1) / thr), thr, 0, stream>>>(qkv, qb, kb, vtb);

    dim3 ga(T_ / 128, B_ * H_);
    k_attn<<<ga, thr, 0, stream>>>(qb, kb, vtb, ob);

    dim3 gp(C_ / 64, M_ / 128);
    k_gemm<<<gp, thr, 0, stream>>>(ob, C_, wpt, C_, bp, h1f, hattn, nullptr, C_, C_, 0);

    k_ln<<<M_, thr, 0, stream>>>(hattn, g2, be2, h2f, h2b, C_);

    dim3 gf1(DFF_ / 64, M_ / 128);
    k_gemm<<<gf1, thr, 0, stream>>>(h2b, C_, w1t, C_, b1, nullptr, nullptr, act, DFF_, C_, 1);

    k_gemm<<<gp, thr, 0, stream>>>(act, DFF_, w2t, DFF_, b2, h2f, (float*)d_out, nullptr, C_, DFF_, 0);
}
